// TransformerEncoderLayer_83038897701239
// MI455X (gfx1250) — matrix operand order repaired and checked by reading, not compiled, not run on hardware
//
#include <hip/hip_runtime.h>
#include <hip/hip_bf16.h>

// ---------------------------------------------------------------------------
// Transformer encoder layer for gfx1250 (MI455X).
// - bf16 WMMA (v_wmma_f32_16x16x32_bf16) for all GEMMs + flash attention
// - global_load_async_to_lds_b128 (ASYNCcnt) with double-buffered LDS tiles
// - fragment loads are 2x ds_load_b128 via layout-aware staging
// B=2, S=2048, HID=1024, H=16, D=64, FF=4096.
// ---------------------------------------------------------------------------

#define BN_ 2
#define SS  2048
#define HIDN 1024
#define NH  16
#define HD  64
#define FF  4096
#define NROWS (BN_ * SS)
#define LN_EPS 1e-5f
#define NEGV  -1000000000.0f

typedef __attribute__((ext_vector_type(16))) __bf16 v16bf;
typedef __attribute__((ext_vector_type(8)))  __bf16 v8bf;
typedef __attribute__((ext_vector_type(8)))  float  v8f;

union Frag16 { v16bf v; v8bf h[2]; };

// Two 16B LDS loads -> one 16x16x32 bf16 fragment.
__device__ inline v16bf ld_frag(const __bf16* p0, const __bf16* p1) {
    Frag16 f;
    f.h[0] = *(const v8bf*)p0;
    f.h[1] = *(const v8bf*)p1;
    return f.v;
}

// Async DMA: 16 bytes global -> LDS per lane, tracked by ASYNCcnt.
// LDS address = low 32 bits of the generic pointer (aperture rule).
__device__ inline void async_cp16(const __bf16* gsrc, __bf16* ldst) {
    const unsigned lds = (unsigned)(uintptr_t)ldst;
    const unsigned long long ga = (unsigned long long)(uintptr_t)gsrc;
    asm volatile("global_load_async_to_lds_b128 %0, %1, off"
                 :: "v"(lds), "v"(ga) : "memory");
}
__device__ inline void wait_async0() {
    asm volatile("s_wait_asynccnt 0x0" ::: "memory");
}
__device__ inline void wait_async4() {
    asm volatile("s_wait_asynccnt 0x4" ::: "memory");
}
__device__ inline void wait_async6() {
    asm volatile("s_wait_asynccnt 0x6" ::: "memory");
}

// ---------------------------------------------------------------------------
// utility kernels
// ---------------------------------------------------------------------------

// LDS-tiled transpose + f32->bf16: src (K x N) -> dst (N x K).
__global__ void transpose_to_bf16(const float* __restrict__ src,
                                  __bf16* __restrict__ dst, int K, int N) {
    __shared__ float tile[32][33];
    const int kb = blockIdx.y * 32, nb = blockIdx.x * 32;
    const int tx = threadIdx.x & 31, ty = threadIdx.x >> 5;
    #pragma unroll
    for (int it = 0; it < 4; ++it) {
        const int i = ty + it * 8;
        tile[i][tx] = src[(size_t)(kb + i) * N + nb + tx];
    }
    __syncthreads();
    #pragma unroll
    for (int it = 0; it < 4; ++it) {
        const int i = ty + it * 8;
        dst[(size_t)(nb + i) * K + kb + tx] = (__bf16)tile[tx][i];
    }
}

// One block per row of 1024; LayerNorm -> bf16 output.
__global__ void layernorm_bf16(const float* __restrict__ x,
                               const float* __restrict__ g,
                               const float* __restrict__ b,
                               __bf16* __restrict__ out) {
    __shared__ float red[256];
    const int row = blockIdx.x;
    const int tid = threadIdx.x;
    const float* xr = x + (size_t)row * HIDN;

    float s = 0.f;
    #pragma unroll
    for (int it = 0; it < 4; ++it) s += xr[tid + it * 256];
    red[tid] = s; __syncthreads();
    for (int off = 128; off > 0; off >>= 1) {
        if (tid < off) red[tid] += red[tid + off];
        __syncthreads();
    }
    const float mean = red[0] * (1.0f / HIDN);
    __syncthreads();

    float v = 0.f;
    #pragma unroll
    for (int it = 0; it < 4; ++it) {
        float d = xr[tid + it * 256] - mean; v += d * d;
    }
    red[tid] = v; __syncthreads();
    for (int off = 128; off > 0; off >>= 1) {
        if (tid < off) red[tid] += red[tid + off];
        __syncthreads();
    }
    const float inv_std = rsqrtf(red[0] * (1.0f / HIDN) + LN_EPS);

    #pragma unroll
    for (int it = 0; it < 4; ++it) {
        const int c = tid + it * 256;
        float y = (xr[c] - mean) * inv_std * g[c] + b[c];
        out[(size_t)row * HIDN + c] = (__bf16)y;
    }
}

// RoPE + reorder (n,hid) -> (b,h,s,d) for Q and K.
__global__ void rope_reorder(const __bf16* __restrict__ qlin,
                             const __bf16* __restrict__ klin,
                             __bf16* __restrict__ qr,
                             __bf16* __restrict__ kr) {
    const int t = blockIdx.x * blockDim.x + threadIdx.x;
    if (t >= NROWS * NH * 32) return;
    const int i = t & 31;
    const int h = (t >> 5) & (NH - 1);
    const int n = t >> 9;
    const int s = n & (SS - 1);
    const int b = n >> 11;

    const float inv_freq = __expf(-((float)i / 32.0f) * 9.210340371976184f);
    float sn, cs; __sincosf((float)s * inv_freq, &sn, &cs);

    const size_t ibase = (size_t)n * HIDN + h * HD;
    const size_t obase = (((size_t)b * NH + h) * SS + s) * HD;
    {
        float e = (float)qlin[ibase + 2 * i], o = (float)qlin[ibase + 2 * i + 1];
        qr[obase + i]      = (__bf16)(e * cs - o * sn);
        qr[obase + i + 32] = (__bf16)(e * sn + o * cs);
    }
    {
        float e = (float)klin[ibase + 2 * i], o = (float)klin[ibase + 2 * i + 1];
        kr[obase + i]      = (__bf16)(e * cs - o * sn);
        kr[obase + i + 32] = (__bf16)(e * sn + o * cs);
    }
}

// V: (n, hid) bf16 -> (b,h,d,s) bf16 (per-head transpose, LDS-tiled).
__global__ void v_transpose(const __bf16* __restrict__ vlin,
                            __bf16* __restrict__ vt) {
    __shared__ __bf16 tile[32][33];
    const int s0 = blockIdx.x * 32, d0 = blockIdx.y * 32, bh = blockIdx.z;
    const int b = bh >> 4, h = bh & (NH - 1);
    const int tx = threadIdx.x & 31, ty = threadIdx.x >> 5;
    #pragma unroll
    for (int it = 0; it < 4; ++it) {
        const int i = ty + it * 8;
        tile[i][tx] = vlin[((size_t)b * SS + s0 + i) * HIDN + h * HD + d0 + tx];
    }
    __syncthreads();
    #pragma unroll
    for (int it = 0; it < 4; ++it) {
        const int i = ty + it * 8;
        vt[((size_t)bh * HD + d0 + i) * SS + s0 + tx] = tile[tx][i];
    }
}

__device__ inline float gelu_tanh(float x) {
    float x3 = x * x * x;
    float t = tanhf(0.7978845608028654f * (x + 0.044715f * x3));
    return 0.5f * x * (1.0f + t);
}

// ---------------------------------------------------------------------------
// Tiled bf16 GEMM: C = A(MxK) * W^T + bias [+ residual] [GELU]
// Weights pre-transposed: Bt is (N x K) row-major.
// Block 256 thr (8 waves), block tile 64x128, wave tile 32x32, K-step 64.
// Double-buffered LDS, async global->LDS DMA overlapped with WMMA.
// ---------------------------------------------------------------------------
#define AST 72
#define BST 72

template <bool OUT_BF16, bool GELU>
__global__ void gemm_bf16_wmma(const __bf16* __restrict__ A,
                               const __bf16* __restrict__ Bt,
                               const float* __restrict__ bias,
                               const float* __restrict__ residual,
                               float* __restrict__ Cf,
                               __bf16* __restrict__ Cb,
                               int M, int N, int K) {
    __shared__ __bf16 As[2][64 * AST];
    __shared__ __bf16 Bs[2][128 * BST];

    const int tid  = threadIdx.x;
    const int lane = tid & 31;
    const int wid  = tid >> 5;
    const int wm = wid >> 2;            // 0..1
    const int wn = wid & 3;             // 0..3
    const int m0 = blockIdx.y * 64;
    const int n0 = blockIdx.x * 128;

    const int mrow = lane & 15;
    const int ncol = lane & 15;
    const int hi   = lane >> 4;

    // One K-tile of async copies: 2 + 4 = 6 wave-instructions (uniform trips).
    auto issue_tile = [&](int buf, int kk) {
        #pragma unroll
        for (int it = 0; it < 2; ++it) {                 // A: 64x64
            const int t = tid + it * 256;
            const int r = t >> 3, c = (t & 7) << 3;
            async_cp16(&A[(size_t)(m0 + r) * K + kk + c], &As[buf][r * AST + c]);
        }
        #pragma unroll
        for (int it = 0; it < 4; ++it) {                 // B: 128x64 (n-major)
            const int t = tid + it * 256;
            const int r = t >> 3, c = (t & 7) << 3;
            async_cp16(&Bt[(size_t)(n0 + r) * K + kk + c], &Bs[buf][r * BST + c]);
        }
    };

    v8f acc[2][2];
    #pragma unroll
    for (int a = 0; a < 2; ++a)
        #pragma unroll
        for (int c = 0; c < 2; ++c) acc[a][c] = (v8f)0.0f;

    const int nk = K >> 6;
    issue_tile(0, 0);
    for (int i = 0; i < nk; ++i) {
        if (i + 1 < nk) {
            issue_tile((i + 1) & 1, (i + 1) << 6);
            wait_async6();               // retire group i, keep group i+1
        } else {
            wait_async0();
        }
        __syncthreads();

        const __bf16* Asb = &As[i & 1][0];
        const __bf16* Bsb = &Bs[i & 1][0];
        #pragma unroll
        for (int kc = 0; kc < 2; ++kc) {
            const int ko = kc * 32;
            v16bf af[2], bfr[2];
            #pragma unroll
            for (int mt = 0; mt < 2; ++mt) {
                const __bf16* pa =
                    &Asb[(wm * 32 + mt * 16 + mrow) * AST + ko + hi * 16];
                af[mt] = ld_frag(pa, pa + 8);
            }
            #pragma unroll
            for (int nt = 0; nt < 2; ++nt) {
                const __bf16* pb =
                    &Bsb[(wn * 32 + nt * 16 + ncol) * BST + ko + hi * 16];
                bfr[nt] = ld_frag(pb, pb + 8);
            }
            #pragma unroll
            for (int mt = 0; mt < 2; ++mt)
                #pragma unroll
                for (int nt = 0; nt < 2; ++nt)
                    acc[mt][nt] = __builtin_amdgcn_wmma_f32_16x16x32_bf16(
                        false, af[mt], false, bfr[nt], (short)0, acc[mt][nt],
                        false, false);
        }
        __syncthreads();                 // buffer free for overwrite
    }

    // Epilogue. C/D layout: lanes 0-15: N=lane, M=vgpr; lanes 16-31: M=8+vgpr.
    const int moff = hi << 3;
    #pragma unroll
    for (int mt = 0; mt < 2; ++mt) {
        #pragma unroll
        for (int nt = 0; nt < 2; ++nt) {
            #pragma unroll
            for (int i = 0; i < 8; ++i) {
                const int row = m0 + wm * 32 + mt * 16 + moff + i;
                const int col = n0 + wn * 32 + nt * 16 + ncol;
                float v = acc[mt][nt][i] + bias[col];
                if (GELU) v = gelu_tanh(v);
                if (residual) v += residual[(size_t)row * N + col];
                if (OUT_BF16) Cb[(size_t)row * N + col] = (__bf16)v;
                else          Cf[(size_t)row * N + col] = v;
            }
        }
    }
}

// ---------------------------------------------------------------------------
// Flash attention: 64 queries x one (b,h) per block; 128 thr (4 waves);
// 32-key chunks (double-buffered async K/V); WMMA Q.K^T and P.V;
// online softmax (lanes 0..15 own rows).
// ---------------------------------------------------------------------------
#define QST 72
#define KST 72
#define VST 48

__global__ void flash_attn(const __bf16* __restrict__ q,   // (b,h,s,d)
                           const __bf16* __restrict__ k,   // (b,h,s,d)
                           const __bf16* __restrict__ vt,  // (b,h,d,s)
                           const int* __restrict__ mask,   // (b,s)
                           __bf16* __restrict__ ctx) {     // (n,hid)
    __shared__ __bf16 Qs[64 * QST];
    __shared__ __bf16 Ks[2][32 * KST];
    __shared__ __bf16 Vts[2][64 * VST];
    __shared__ float  Sc[4][16 * 32];
    __shared__ __bf16 Pb[4][16 * 32];
    __shared__ float  Al[4 * 16];

    const int tid  = threadIdx.x;        // 0..127
    const int lane = tid & 31;
    const int w    = tid >> 5;
    const int qt = blockIdx.x, h = blockIdx.y, b = blockIdx.z;
    const int q0 = qt * 64;
    const int bh = b * NH + h;

    const size_t head_base = (size_t)bh * SS * HD;     // q, k
    const size_t vhead     = (size_t)bh * HD * SS;     // vt

    const int mrow = lane & 15;
    const int ncol = lane & 15;
    const int hi   = lane >> 4;
    const int moff = hi << 3;

    // K/V chunk copies: 2 + 2 = 4 wave-instructions (uniform trips).
    auto issue_kv = [&](int buf, int kc) {
        #pragma unroll
        for (int it = 0; it < 2; ++it) {                 // K: 32x64
            const int t = tid + it * 128;
            const int r = t >> 3, c = (t & 7) << 3;
            async_cp16(&k[head_base + (size_t)(kc + r) * HD + c],
                       &Ks[buf][r * KST + c]);
        }
        #pragma unroll
        for (int it = 0; it < 2; ++it) {                 // Vt: 64x32
            const int t = tid + it * 128;
            const int r = t >> 2, c = (t & 3) << 3;
            async_cp16(&vt[vhead + (size_t)r * SS + kc + c],
                       &Vts[buf][r * VST + c]);
        }
    };

    // Q tile (64x64): async, 4 wave-instructions.
    #pragma unroll
    for (int it = 0; it < 4; ++it) {
        const int t = tid + it * 128;
        const int r = t >> 3, c = (t & 7) << 3;
        async_cp16(&q[head_base + (size_t)(q0 + r) * HD + c], &Qs[r * QST + c]);
    }
    issue_kv(0, 0);
    wait_async4();                       // Q done; kv0 still in flight
    __syncthreads();

    v16bf aQ[2];
    #pragma unroll
    for (int dc = 0; dc < 2; ++dc) {
        const __bf16* pa = &Qs[(w * 16 + mrow) * QST + dc * 32 + hi * 16];
        aQ[dc] = ld_frag(pa, pa + 8);
    }

    v8f acc[4];
    #pragma unroll
    for (int dt = 0; dt < 4; ++dt) acc[dt] = (v8f)0.0f;
    float run_max = -3.0e38f, run_sum = 0.0f;   // lanes 0..15

    const int nchunks = 2 * (qt + 1);
    for (int ch = 0; ch < nchunks; ++ch) {
        const int kc = ch * 32;
        if (ch + 1 < nchunks) {
            issue_kv((ch + 1) & 1, kc + 32);
            wait_async4();               // retire chunk ch
        } else {
            wait_async0();
        }
        __syncthreads();
        const __bf16* Ksb = &Ks[ch & 1][0];
        const __bf16* Vsb = &Vts[ch & 1][0];

        // scores: S(16x32) = Q . K^T
        v8f sf[2] = {(v8f)0.0f, (v8f)0.0f};
        #pragma unroll
        for (int dc = 0; dc < 2; ++dc) {
            #pragma unroll
            for (int nt = 0; nt < 2; ++nt) {
                const __bf16* pb = &Ksb[(nt * 16 + ncol) * KST + dc * 32 + hi * 16];
                sf[nt] = __builtin_amdgcn_wmma_f32_16x16x32_bf16(
                    false, aQ[dc], false, ld_frag(pb, pb + 8), (short)0,
                    sf[nt], false, false);
            }
        }

        // scale + causal/pad mask -> per-wave LDS score tile
        #pragma unroll
        for (int nt = 0; nt < 2; ++nt) {
            const int key_g = kc + nt * 16 + ncol;
            const int mk = mask[(size_t)b * SS + key_g];
            #pragma unroll
            for (int i = 0; i < 8; ++i) {
                const int qm = moff + i;
                const int q_g = q0 + w * 16 + qm;
                float s = sf[nt][i] * 0.125f;
                if (key_g > q_g || mk == 0) s = NEGV;
                Sc[w][qm * 32 + nt * 16 + ncol] = s;
            }
        }
        __syncthreads();

        // online softmax: lanes 0..15 each own one row of 32 scores
        if (lane < 16) {
            const float* row = &Sc[w][lane * 32];
            float mx = run_max;
            #pragma unroll
            for (int c = 0; c < 32; ++c) mx = fmaxf(mx, row[c]);
            const float alpha = __expf(run_max - mx);
            float sum = 0.f;
            #pragma unroll
            for (int c = 0; c < 32; ++c) {
                float p = __expf(row[c] - mx);
                sum += p;
                Pb[w][lane * 32 + c] = (__bf16)p;
            }
            run_sum = run_sum * alpha + sum;
            run_max = mx;
            Al[w * 16 + lane] = alpha;
        }
        __syncthreads();

        // rescale accumulators, then ctx += P(16x32) . V(32x64)
        float al[8];
        #pragma unroll
        for (int i = 0; i < 8; ++i) al[i] = Al[w * 16 + moff + i];
        #pragma unroll
        for (int dt = 0; dt < 4; ++dt)
            #pragma unroll
            for (int i = 0; i < 8; ++i) acc[dt][i] *= al[i];

        const __bf16* pp = &Pb[w][mrow * 32 + hi * 16];
        const v16bf aP = ld_frag(pp, pp + 8);
        #pragma unroll
        for (int dt = 0; dt < 4; ++dt) {
            const __bf16* pv = &Vsb[(dt * 16 + ncol) * VST + hi * 16];
            acc[dt] = __builtin_amdgcn_wmma_f32_16x16x32_bf16(
                false, aP, false, ld_frag(pv, pv + 8), (short)0, acc[dt],
                false, false);
        }
        __syncthreads();                 // chunk buffers free for overwrite
    }

    // normalize and store to ctx (n, hid), hid = h*64 + d
    if (lane < 16) Al[w * 16 + lane] = run_sum;
    __syncthreads();
    #pragma unroll
    for (int i = 0; i < 8; ++i) {
        const int qm = moff + i;
        const float inv = 1.0f / Al[w * 16 + qm];
        const size_t nrow = (size_t)b * SS + q0 + w * 16 + qm;
        #pragma unroll
        for (int dt = 0; dt < 4; ++dt) {
            const int col = h * HD + dt * 16 + ncol;
            ctx[nrow * HIDN + col] = (__bf16)(acc[dt][i] * inv);
        }
    }
}

// ---------------------------------------------------------------------------
// host: workspace carve + launch sequence
// ---------------------------------------------------------------------------
extern "C" void kernel_launch(void* const* d_in, const int* in_sizes, int n_in,
                              void* d_out, int out_size, void* d_ws, size_t ws_size,
                              hipStream_t stream) {
    const float* x    = (const float*)d_in[0];
    const int*   mask = (const int*)  d_in[1];
    const float* wq = (const float*)d_in[2];  const float* bq = (const float*)d_in[3];
    const float* wk = (const float*)d_in[4];  const float* bk = (const float*)d_in[5];
    const float* wv = (const float*)d_in[6];  const float* bv = (const float*)d_in[7];
    const float* wo = (const float*)d_in[8];  const float* bo = (const float*)d_in[9];
    const float* w1 = (const float*)d_in[10]; const float* b1 = (const float*)d_in[11];
    const float* w2 = (const float*)d_in[12]; const float* b2 = (const float*)d_in[13];
    const float* ln1g = (const float*)d_in[14]; const float* ln1b = (const float*)d_in[15];
    const float* ln2g = (const float*)d_in[16]; const float* ln2b = (const float*)d_in[17];
    float* out = (float*)d_out;

    char* p = (char*)d_ws;
    auto alloc = [&](size_t bytes) -> void* {
        void* r = (void*)p;
        p += (bytes + 255) & ~(size_t)255;
        return r;
    };
    const size_t NH1 = (size_t)NROWS * HIDN;
    __bf16* xn_bf = (__bf16*)alloc(NH1 * 2);
    __bf16* wqb   = (__bf16*)alloc((size_t)HIDN * HIDN * 2);   // (N x K)
    __bf16* wkb   = (__bf16*)alloc((size_t)HIDN * HIDN * 2);
    __bf16* wvb   = (__bf16*)alloc((size_t)HIDN * HIDN * 2);
    __bf16* wob   = (__bf16*)alloc((size_t)HIDN * HIDN * 2);
    __bf16* w1b   = (__bf16*)alloc((size_t)HIDN * FF * 2);     // (FF x HIDN)
    __bf16* w2b   = (__bf16*)alloc((size_t)FF * HIDN * 2);     // (HIDN x FF)
    __bf16* qlin  = (__bf16*)alloc(NH1 * 2);
    __bf16* klin  = (__bf16*)alloc(NH1 * 2);
    __bf16* vlin  = (__bf16*)alloc(NH1 * 2);
    __bf16* qr    = (__bf16*)alloc(NH1 * 2);
    __bf16* kr    = (__bf16*)alloc(NH1 * 2);
    __bf16* vtb   = (__bf16*)alloc(NH1 * 2);                   // (b,h,d,s)
    __bf16* ctxb  = (__bf16*)alloc(NH1 * 2);
    __bf16* xn2b  = (__bf16*)alloc(NH1 * 2);
    __bf16* ffhb  = (__bf16*)alloc((size_t)NROWS * FF * 2);

    // 1) weights -> bf16, transposed to (N x K)
    dim3 gHH(HIDN / 32, HIDN / 32);
    transpose_to_bf16<<<gHH, 256, 0, stream>>>(wq, wqb, HIDN, HIDN);
    transpose_to_bf16<<<gHH, 256, 0, stream>>>(wk, wkb, HIDN, HIDN);
    transpose_to_bf16<<<gHH, 256, 0, stream>>>(wv, wvb, HIDN, HIDN);
    transpose_to_bf16<<<gHH, 256, 0, stream>>>(wo, wob, HIDN, HIDN);
    dim3 gW1(FF / 32, HIDN / 32);
    transpose_to_bf16<<<gW1, 256, 0, stream>>>(w1, w1b, HIDN, FF);
    dim3 gW2(HIDN / 32, FF / 32);
    transpose_to_bf16<<<gW2, 256, 0, stream>>>(w2, w2b, FF, HIDN);

    // 2) LN1
    layernorm_bf16<<<NROWS, 256, 0, stream>>>(x, ln1g, ln1b, xn_bf);

    // 3) QKV projections (M=4096, N=1024, K=1024)
    dim3 gP(HIDN / 128, NROWS / 64);
    gemm_bf16_wmma<true, false><<<gP, 256, 0, stream>>>(
        xn_bf, wqb, bq, nullptr, nullptr, qlin, NROWS, HIDN, HIDN);
    gemm_bf16_wmma<true, false><<<gP, 256, 0, stream>>>(
        xn_bf, wkb, bk, nullptr, nullptr, klin, NROWS, HIDN, HIDN);
    gemm_bf16_wmma<true, false><<<gP, 256, 0, stream>>>(
        xn_bf, wvb, bv, nullptr, nullptr, vlin, NROWS, HIDN, HIDN);

    // 4) RoPE (q,k) and per-head V transpose
    rope_reorder<<<(NROWS * NH * 32) / 256, 256, 0, stream>>>(qlin, klin, qr, kr);
    dim3 gV(SS / 32, HD / 32, BN_ * NH);
    v_transpose<<<gV, 256, 0, stream>>>(vlin, vtb);

    // 5) attention
    dim3 gA(SS / 64, NH, BN_);
    flash_attn<<<gA, 128, 0, stream>>>(qr, kr, vtb, mask, ctxb);

    // 6) out projection + residual(x) -> d_out (f32)
    gemm_bf16_wmma<false, false><<<gP, 256, 0, stream>>>(
        ctxb, wob, bo, x, out, nullptr, NROWS, HIDN, HIDN);

    // 7) LN2 on d_out
    layernorm_bf16<<<NROWS, 256, 0, stream>>>(out, ln2g, ln2b, xn2b);

    // 8) FFN1 + GELU (M=4096, N=4096, K=1024)
    dim3 gF1(FF / 128, NROWS / 64);
    gemm_bf16_wmma<true, true><<<gF1, 256, 0, stream>>>(
        xn2b, w1b, b1, nullptr, nullptr, ffhb, NROWS, FF, HIDN);

    // 9) FFN2 + residual(d_out) -> d_out (M=4096, N=1024, K=4096)
    gemm_bf16_wmma<false, false><<<gP, 256, 0, stream>>>(
        ffhb, w2b, b2, out, out, nullptr, NROWS, HIDN, FF);
}
